// SwinTransformerBlock_16398185136203
// MI455X (gfx1250) — compile-verified, matrix operand order checked by reading, not run on hardware
//
#include <hip/hip_runtime.h>
#include <math.h>

// ---------------------------------------------------------------------------
// MI455X (gfx1250) Swin 3D block. All GEMMs via v_wmma_f32_16x16x32_bf16.
// wave32, 256 threads/block (8 waves). Dynamic LDS keeps each window resident.
// ---------------------------------------------------------------------------

typedef __attribute__((ext_vector_type(16))) __bf16 v16bf;
typedef __attribute__((ext_vector_type(8)))  __bf16 v8bf;
typedef __attribute__((ext_vector_type(8)))  float  v8f;

#define C_DIM   192
#define NHEAD   6
#define HDIM    32
#define NTOK    98      // tokens per window (2*7*7)
#define NPAD    112     // padded to 7 tiles of 16
#define KVPAD   128     // K padding for P@V (4 tiles of 32)
#define NN      (NTOK * NTOK)

// WMMA bf16 operand load: row-major matrix, A/B layout per CDNA5 ISA 7.12.2.
// lanes 0-15 : row = base+lane,    elems 0..7 = K+0..7,  elems 8..15 = K+16..23
// lanes 16-31: row = base+lane-16, K offset +8 -> K+8..15 and K+24..31
__device__ __forceinline__ v16bf load_op(const __bf16* p) {
  v8bf lo = *(const v8bf*)p;
  v8bf hi = *(const v8bf*)(p + 16);
  v16bf r;
#pragma unroll
  for (int i = 0; i < 8; ++i) { r[i] = lo[i]; r[i + 8] = hi[i]; }
  return r;
}

__device__ __forceinline__ const __bf16* opp(const __bf16* base, int ld,
                                             int rowBase, int kBase, int lane) {
  return base + (size_t)(rowBase + (lane & 15)) * ld + kBase + ((lane & 16) ? 8 : 0);
}

__device__ __forceinline__ v8f wmma_bf16(v16bf a, v16bf b, v8f c) {
  return __builtin_amdgcn_wmma_f32_16x16x32_bf16(false, a, false, b, (short)0, c,
                                                 false, false);
}

__device__ __forceinline__ float wred_sum(float v) {
#pragma unroll
  for (int m = 16; m >= 1; m >>= 1) v += __shfl_xor(v, m, 32);
  return v;
}
__device__ __forceinline__ float wred_max(float v) {
#pragma unroll
  for (int m = 16; m >= 1; m >>= 1) v = fmaxf(v, __shfl_xor(v, m, 32));
  return v;
}

// fp32 -> bf16 weight conversion (weights then live in L2: < 1 MB total)
__global__ void cvt_bf16_kernel(const float* __restrict__ src,
                                __bf16* __restrict__ dst, int n) {
  int i = blockIdx.x * blockDim.x + threadIdx.x;
  if (i < n) dst[i] = (__bf16)src[i];
}

// Pre-gather rel-pos bias per head: biasH[h][i][j] = rpb[rel_index[i,j], h]
// Removes the dependent 2-level gather from the attention score epilogue.
__global__ void bias_combine_kernel(const int* __restrict__ relidx,
                                    const float* __restrict__ rpb,
                                    float* __restrict__ biasH, int n) {
  int i = blockIdx.x * blockDim.x + threadIdx.x;
  if (i < n) {
    int h = i / NN, ij = i % NN;
    biasH[i] = rpb[relidx[ij] * NHEAD + h];
  }
}

// ---------------------------------------------------------------------------
// Mega kernel: one workgroup per shifted window.
// LDS: xw/ao [112][192]bf16 | qk [112][384]bf16 | vT [192][128]bf16 |
//      S [112][112]f32      | P  [112][128]bf16 | gmap[112]i32      = 251 KB
// ---------------------------------------------------------------------------
#define OFF_QK   43008
#define OFF_VT   129024
#define OFF_S    178176
#define OFF_P    228352
#define OFF_GMAP 257024
#define SMEM_ATTN 257472

__global__ __launch_bounds__(256) void swin_attn_kernel(
    const float* __restrict__ x, const float* __restrict__ mask,
    const float* __restrict__ biasH, const float* __restrict__ n1g,
    const float* __restrict__ n1b, const __bf16* __restrict__ wqkv,
    const float* __restrict__ qkvb, const __bf16* __restrict__ wproj,
    const float* __restrict__ projb, float* __restrict__ out) {
  extern __shared__ char smem[];
  __bf16* xw = (__bf16*)smem;               // LN'd window tokens, later attn out
  __bf16* qk = (__bf16*)(smem + OFF_QK);    // cols 0..191 = q*scale, 192..383 = k
  __bf16* vt = (__bf16*)(smem + OFF_VT);    // v transposed: [NH*32][128 tokens]
  float*  S  = (float*)(smem + OFF_S);      // scores per head
  __bf16* P  = (__bf16*)(smem + OFF_P);     // softmax probs (K-padded to 128)
  int*    gmap = (int*)(smem + OFF_GMAP);   // token -> flat spatial index

  const int tid  = threadIdx.x;
  const int lane = tid & 31;
  const int wave = tid >> 5;
  const int win  = blockIdx.x;
  const int b    = win >> 8;       // 256 windows per batch element
  const int wI   = win & 255;      // window index within batch (mask index)
  const int wdi = wI >> 6, whi = (wI >> 3) & 7, wwi = wI & 7;

  // zero vT so K-pad columns (112..127) contribute exactly 0 in P@V
  {
    uint4* vz = (uint4*)vt;
#pragma unroll 2
    for (int i = tid; i < (192 * KVPAD * 2) / 16; i += 256)
      vz[i] = make_uint4(0u, 0u, 0u, 0u);
  }

  // ---- Phase 0: LN1 + cyclic shift + window partition gather -> xw (bf16)
  for (int t = wave; t < NPAD; t += 8) {
    if (t < NTOK) {
      int td = t / 49, rh = t % 49;
      int th = rh / 7, tw = rh % 7;
      int dsh = wdi * 2 + td, hsh = whi * 7 + th, wsh = wwi * 7 + tw;
      int d0 = (dsh + 1) & 7;                       // roll(-1) over D=8
      int h0 = hsh + 3; if (h0 >= 56) h0 -= 56;     // roll(-3) over H=56
      int w0 = wsh + 3; if (w0 >= 56) w0 -= 56;     // roll(-3) over W=56
      int g = ((b * 8 + d0) * 56 + h0) * 56 + w0;
      if (lane == 0) gmap[t] = g;
      const float* xr = x + (size_t)g * C_DIM;
      float v0[6]; float s = 0.f, ss = 0.f;
#pragma unroll
      for (int j = 0; j < 6; ++j) {
        float f = xr[lane + 32 * j]; v0[j] = f; s += f; ss += f * f;
      }
      s = wred_sum(s); ss = wred_sum(ss);
      float mu = s * (1.f / 192.f);
      float var = ss * (1.f / 192.f) - mu * mu;
      float rstd = rsqrtf(var + 1e-5f);
#pragma unroll
      for (int j = 0; j < 6; ++j) {
        int c = lane + 32 * j;
        xw[t * C_DIM + c] = (__bf16)((v0[j] - mu) * rstd * n1g[c] + n1b[c]);
      }
    } else {
      if (lane == 0) gmap[t] = -1;
#pragma unroll
      for (int j = 0; j < 6; ++j) xw[t * C_DIM + lane + 32 * j] = (__bf16)0.f;
    }
  }
  __syncthreads();

  // ---- Phase 1: QKV GEMM  [112x192] x [192x576]^T  (WMMA bf16)
  const float qscale = 0.17677669529663687f;  // 1/sqrt(32)
  for (int tile = wave; tile < 7 * 36; tile += 8) {
    int mt = tile / 36, nt = tile % 36;
    int nrow = nt * 16 + (lane & 15);           // output feature (weight row)
    const __bf16* wrow = wqkv + (size_t)nrow * C_DIM + ((lane & 16) ? 8 : 0);
    __builtin_prefetch(wrow + 128 * C_DIM, 0, 0);  // wave's next tile (L2 hint)
    v8f acc = {0.f, 0.f, 0.f, 0.f, 0.f, 0.f, 0.f, 0.f};
#pragma unroll
    for (int kt = 0; kt < 6; ++kt) {
      v16bf a  = load_op(opp(xw, C_DIM, mt * 16, kt * 32, lane));
      v16bf bm = load_op(wrow + kt * 32);
      acc = wmma_bf16(a, bm, acc);
    }
    float bias = qkvb[nrow];
    int hi = (lane & 16) ? 8 : 0;
    // q/k/v regions align to 16-wide N tiles -> wave-uniform branch on nt
    if (nt < 12) {                                  // q (pre-scaled)
#pragma unroll
      for (int r = 0; r < 8; ++r)
        qk[(mt * 16 + hi + r) * 384 + nrow] = (__bf16)((acc[r] + bias) * qscale);
    } else if (nt < 24) {                           // k
#pragma unroll
      for (int r = 0; r < 8; ++r)
        qk[(mt * 16 + hi + r) * 384 + nrow] = (__bf16)(acc[r] + bias);
    } else {                                        // v, stored transposed
      int vr = nrow - 384;
#pragma unroll
      for (int r = 0; r < 8; ++r)
        vt[vr * KVPAD + mt * 16 + hi + r] = (__bf16)(acc[r] + bias);
    }
  }
  __syncthreads();

  // ---- Phase 2: per-head attention, fully in LDS
  for (int h = 0; h < NHEAD; ++h) {
    const float* bH = biasH + (size_t)h * NN;
    const float* mW = mask + (size_t)wI * NN;
    // 2a: S = q_h @ k_h^T  + rel-pos bias + shift mask  (49 tiles, K=32)
    for (int tile = wave; tile < 49; tile += 8) {
      int mt = tile / 7, nt = tile % 7;
      v8f acc = {0.f, 0.f, 0.f, 0.f, 0.f, 0.f, 0.f, 0.f};
      v16bf a  = load_op(opp(qk, 384, mt * 16, h * 32, lane));
      v16bf bm = load_op(opp(qk, 384, nt * 16, 192 + h * 32, lane));
      acc = wmma_bf16(a, bm, acc);
      int hi = (lane & 16) ? 8 : 0;
      int j = nt * 16 + (lane & 15);
#pragma unroll
      for (int r = 0; r < 8; ++r) {
        int i = mt * 16 + hi + r;
        float v = acc[r];
        if (j < NTOK) {
          if (i < NTOK) v += bH[i * NTOK + j] + mW[i * NTOK + j];
        } else {
          v = -1e30f;  // mask padded key columns
        }
        S[i * NPAD + j] = v;
      }
    }
    __syncthreads();

    // 2b: row softmax, write bf16 probs (cols 112..127 zeroed)
    for (int row = wave; row < NPAD; row += 8) {
      const float* sr = S + row * NPAD;
      bool has4 = lane < 16;                 // 112 = 3*32 + 16
      float s0 = sr[lane];
      float s1 = sr[lane + 32];
      float s2 = sr[lane + 64];
      float s3 = has4 ? sr[lane + 96] : -1e30f;
      float m = wred_max(fmaxf(fmaxf(s0, s1), fmaxf(s2, s3)));
      float e0 = __expf(s0 - m), e1 = __expf(s1 - m), e2 = __expf(s2 - m);
      float e3 = has4 ? __expf(s3 - m) : 0.f;
      float inv = 1.f / wred_sum(e0 + e1 + e2 + e3);
      __bf16* pr = P + row * KVPAD;
      pr[lane]      = (__bf16)(e0 * inv);
      pr[lane + 32] = (__bf16)(e1 * inv);
      pr[lane + 64] = (__bf16)(e2 * inv);
      if (has4) {
        pr[lane + 96]  = (__bf16)(e3 * inv);
        pr[lane + 112] = (__bf16)0.f;        // K-pad columns
      }
    }
    __syncthreads();

    // 2c: out_h = P @ v_h  (A = probs [112x128], B = vT rows, K = 4x32)
    for (int tile = wave; tile < 14; tile += 8) {
      int mt = tile / 2, nt = tile % 2;
      v8f acc = {0.f, 0.f, 0.f, 0.f, 0.f, 0.f, 0.f, 0.f};
#pragma unroll
      for (int kt = 0; kt < 4; ++kt) {
        v16bf a  = load_op(opp(P, KVPAD, mt * 16, kt * 32, lane));
        v16bf bm = load_op(opp(vt, KVPAD, h * 32 + nt * 16, kt * 32, lane));
        acc = wmma_bf16(a, bm, acc);
      }
      int hi = (lane & 16) ? 8 : 0;
      int col = h * 32 + nt * 16 + (lane & 15);
#pragma unroll
      for (int r = 0; r < 8; ++r)
        xw[(mt * 16 + hi + r) * C_DIM + col] = (__bf16)acc[r];  // reuse xw as ao
    }
    __syncthreads();
  }

  // ---- Phase 3: proj GEMM + window reverse + roll back + residual -> out
  for (int tile = wave; tile < 7 * 12; tile += 8) {
    int mt = tile / 12, nt = tile % 12;
    int nrow = nt * 16 + (lane & 15);
    const __bf16* wrow = wproj + (size_t)nrow * C_DIM + ((lane & 16) ? 8 : 0);
    v8f acc = {0.f, 0.f, 0.f, 0.f, 0.f, 0.f, 0.f, 0.f};
#pragma unroll
    for (int kt = 0; kt < 6; ++kt) {
      v16bf a  = load_op(opp(xw, C_DIM, mt * 16, kt * 32, lane));
      v16bf bm = load_op(wrow + kt * 32);
      acc = wmma_bf16(a, bm, acc);
    }
    float bias = projb[nrow];
    int hi = (lane & 16) ? 8 : 0;
    if (mt < 6) {  // rows < 96: always real tokens, no pad check needed
#pragma unroll
      for (int r = 0; r < 8; ++r) {
        size_t off = (size_t)gmap[mt * 16 + hi + r] * C_DIM + nrow;
        out[off] = x[off] + acc[r] + bias;   // x1 = shortcut + attn branch
      }
    } else {       // last M tile contains padded rows 98..111
#pragma unroll
      for (int r = 0; r < 8; ++r) {
        int g = gmap[mt * 16 + hi + r];
        if (g >= 0) {
          size_t off = (size_t)g * C_DIM + nrow;
          out[off] = x[off] + acc[r] + bias;
        }
      }
    }
  }
}

// ---------------------------------------------------------------------------
// Fused MLP: LN2 -> fc1 (WMMA) -> exact GELU -> fc2 (WMMA) -> residual accum.
// 64 tokens per block; LDS: A [64][192]bf16 + H [64][768]bf16 = 120 KB
// ---------------------------------------------------------------------------
#define SMEM_MLP (24576 + 98304)

__global__ __launch_bounds__(256) void swin_mlp_kernel(
    float* __restrict__ io, const float* __restrict__ n2g,
    const float* __restrict__ n2b, const __bf16* __restrict__ wfc1,
    const float* __restrict__ b1, const __bf16* __restrict__ wfc2,
    const float* __restrict__ b2) {
  extern __shared__ char smem[];
  __bf16* A = (__bf16*)smem;             // LN'd tokens [64][192]
  __bf16* H = (__bf16*)(smem + 24576);   // GELU hidden [64][768]
  const int tid = threadIdx.x, lane = tid & 31, wave = tid >> 5;
  const size_t rowbase = (size_t)blockIdx.x * 64;

  // LN2
  for (int t = wave; t < 64; t += 8) {
    const float* xr = io + (rowbase + t) * C_DIM;
    float v0[6]; float s = 0.f, ss = 0.f;
#pragma unroll
    for (int j = 0; j < 6; ++j) {
      float f = xr[lane + 32 * j]; v0[j] = f; s += f; ss += f * f;
    }
    s = wred_sum(s); ss = wred_sum(ss);
    float mu = s * (1.f / 192.f);
    float var = ss * (1.f / 192.f) - mu * mu;
    float rstd = rsqrtf(var + 1e-5f);
#pragma unroll
    for (int j = 0; j < 6; ++j) {
      int c = lane + 32 * j;
      A[t * C_DIM + c] = (__bf16)((v0[j] - mu) * rstd * n2g[c] + n2b[c]);
    }
  }
  __syncthreads();

  // fc1 + exact GELU   (4 M-tiles x 48 N-tiles, K = 6x32)
  for (int tile = wave; tile < 4 * 48; tile += 8) {
    int mt = tile / 48, nt = tile % 48;
    int nrow = nt * 16 + (lane & 15);
    const __bf16* wrow = wfc1 + (size_t)nrow * C_DIM + ((lane & 16) ? 8 : 0);
    __builtin_prefetch(wrow + 128 * C_DIM, 0, 0);  // wave's next tile (L2 hint)
    v8f acc = {0.f, 0.f, 0.f, 0.f, 0.f, 0.f, 0.f, 0.f};
#pragma unroll
    for (int kt = 0; kt < 6; ++kt) {
      v16bf a  = load_op(opp(A, C_DIM, mt * 16, kt * 32, lane));
      v16bf bm = load_op(wrow + kt * 32);
      acc = wmma_bf16(a, bm, acc);
    }
    float bias = b1[nrow];
    int hi = (lane & 16) ? 8 : 0;
#pragma unroll
    for (int r = 0; r < 8; ++r) {
      float v = acc[r] + bias;
      float g = 0.5f * v * (1.f + erff(v * 0.70710678118654752f));
      H[(mt * 16 + hi + r) * 768 + nrow] = (__bf16)g;
    }
  }
  __syncthreads();

  // fc2 + residual accumulate   (4 M-tiles x 12 N-tiles, K = 24x32)
  for (int tile = wave; tile < 4 * 12; tile += 8) {
    int mt = tile / 12, nt = tile % 12;
    int nrow = nt * 16 + (lane & 15);
    const __bf16* wrow = wfc2 + (size_t)nrow * 768 + ((lane & 16) ? 8 : 0);
    __builtin_prefetch(wrow + 128 * 768, 0, 0);    // wave's next tile (L2 hint)
    v8f acc = {0.f, 0.f, 0.f, 0.f, 0.f, 0.f, 0.f, 0.f};
#pragma unroll
    for (int kt = 0; kt < 24; ++kt) {
      v16bf a  = load_op(opp(H, 768, mt * 16, kt * 32, lane));
      v16bf bm = load_op(wrow + kt * 32);
      acc = wmma_bf16(a, bm, acc);
    }
    float bias = b2[nrow];
    int hi = (lane & 16) ? 8 : 0;
#pragma unroll
    for (int r = 0; r < 8; ++r) {
      float* o = io + (rowbase + mt * 16 + hi + r) * C_DIM + nrow;
      *o = *o + acc[r] + bias;
    }
  }
}

// ---------------------------------------------------------------------------
extern "C" void kernel_launch(void* const* d_in, const int* in_sizes, int n_in,
                              void* d_out, int out_size, void* d_ws,
                              size_t ws_size, hipStream_t stream) {
  (void)in_sizes; (void)n_in; (void)out_size; (void)ws_size;
  const float* x      = (const float*)d_in[0];
  const float* amask  = (const float*)d_in[1];
  const int*   relidx = (const int*)d_in[2];
  const float* n1g    = (const float*)d_in[3];
  const float* n1b    = (const float*)d_in[4];
  const float* qkv_w  = (const float*)d_in[5];
  const float* qkv_b  = (const float*)d_in[6];
  const float* rpb    = (const float*)d_in[7];
  const float* proj_w = (const float*)d_in[8];
  const float* proj_b = (const float*)d_in[9];
  const float* n2g    = (const float*)d_in[10];
  const float* n2b    = (const float*)d_in[11];
  const float* fc1_w  = (const float*)d_in[12];
  const float* fc1_b  = (const float*)d_in[13];
  const float* fc2_w  = (const float*)d_in[14];
  const float* fc2_b  = (const float*)d_in[15];
  float* out = (float*)d_out;

  // Workspace: bf16 weight copies + pre-gathered per-head bias (~1.1 MB, L2)
  __bf16* wqkv  = (__bf16*)d_ws;                   // 576*192
  __bf16* wproj = wqkv  + 576 * 192;               // 192*192
  __bf16* wfc1  = wproj + 192 * 192;               // 768*192
  __bf16* wfc2  = wfc1  + 768 * 192;               // 192*768
  float*  biasH = (float*)(wfc2 + 192 * 768);      // [6][98][98]

  cvt_bf16_kernel<<<(576 * 192 + 255) / 256, 256, 0, stream>>>(qkv_w, wqkv, 576 * 192);
  cvt_bf16_kernel<<<(192 * 192 + 255) / 256, 256, 0, stream>>>(proj_w, wproj, 192 * 192);
  cvt_bf16_kernel<<<(768 * 192 + 255) / 256, 256, 0, stream>>>(fc1_w, wfc1, 768 * 192);
  cvt_bf16_kernel<<<(192 * 768 + 255) / 256, 256, 0, stream>>>(fc2_w, wfc2, 192 * 768);
  bias_combine_kernel<<<(NHEAD * NN + 255) / 256, 256, 0, stream>>>(
      relidx, rpb, biasH, NHEAD * NN);

  // 1024 shifted windows, one workgroup each; writes x1 into d_out
  swin_attn_kernel<<<1024, 256, SMEM_ATTN, stream>>>(
      x, amask, biasH, n1g, n1b, wqkv, qkv_b, wproj, proj_b, out);

  // 100352 tokens / 64 per block; accumulates MLP branch into d_out
  swin_mlp_kernel<<<100352 / 64, 256, SMEM_MLP, stream>>>(
      out, n2g, n2b, wfc1, fc1_b, wfc2, fc2_b);
}
